// NATBlock_1769526526662
// MI455X (gfx1250) — compile-verified, matrix operand order checked by reading, not run on hardware
//
#include <hip/hip_runtime.h>
#include <hip/hip_bf16.h>
#include <math.h>

typedef _Float16 half_t;
typedef __attribute__((ext_vector_type(16))) _Float16 v16h;
typedef __attribute__((ext_vector_type(8)))  _Float16 v8h;
typedef __attribute__((ext_vector_type(8)))  float    v8f;

#define B_IMG 2
#define H_IMG 56
#define W_IMG 56
#define C_DIM 128
#define NPIX  (B_IMG * H_IMG * W_IMG)   // 6272
#define NHEAD 4
#define HDIM  32
#define KWIN  7
#define NNB   (KWIN * KWIN)             // 49
#define RPB_D (2 * KWIN - 1)            // 13
#define OH    28
#define OW    28
#define NOUT  (B_IMG * OH * OW)         // 1568
#define KCONV (9 * C_DIM)               // 1152
#define MCONV (2 * C_DIM)               // 256
#define HID   (4 * C_DIM)               // 512

// ---------------------------------------------------------------------------
// Wave reductions (wave32)
// ---------------------------------------------------------------------------
__device__ inline float wave_sum(float v) {
    for (int o = 16; o; o >>= 1) v += __shfl_xor(v, o, 32);
    return v;
}
__device__ inline float wave_max(float v) {
    for (int o = 16; o; o >>= 1) v = fmaxf(v, __shfl_xor(v, o, 32));
    return v;
}

// ---------------------------------------------------------------------------
// WMMA 16-bit fragment loader.
// A (16x32, row-major, ld = K):  lane<16 holds row (lane), K = [k0..k0+7, k0+16..k0+23]
//                                lane>=16 holds row (lane-16), K = [k0+8..k0+15, k0+24..k0+31]
// B loaded identically from the pre-transposed (MxK) weight -> column fragment.
// ---------------------------------------------------------------------------
__device__ inline v16h load_frag16(const half_t* __restrict__ base, int row, int ldk,
                                   int k0, int lane) {
    const half_t* p = base + (size_t)row * ldk + k0 + ((lane >> 4) << 3);
    v8h lo = *(const v8h*)(p);
    v8h hi = *(const v8h*)(p + 16);
    v16h f;
#pragma unroll
    for (int i = 0; i < 8; ++i) { f[i] = lo[i]; f[i + 8] = hi[i]; }
    return f;
}

// ---------------------------------------------------------------------------
// Generic WMMA GEMM: C[N x M] = act(A[N x K] (f16) * Bt[M x K]^T (f16) + bias)
// Block = 256 threads = 8 waves (2x4) -> 64x128 block tile.
// Each wave register-blocks a 32x32 output tile: 2 A-frags x 2 B-frags ->
// 4 accumulators, 4 WMMAs per K-step (2 b128 loads per WMMA, L2-resident).
// OMODE: 0 = store f32, 1 = accumulate into f32 (residual), 2 = store f16.
// ---------------------------------------------------------------------------
template <bool GELU, int OMODE>
__global__ __launch_bounds__(256) void gemm_wmma_kernel(
    const half_t* __restrict__ A, const half_t* __restrict__ Bt,
    const float* __restrict__ bias, float* __restrict__ Cf32,
    half_t* __restrict__ Cf16, int N, int K, int M)
{
    const int tid  = threadIdx.x;
    const int lane = tid & 31;
    const int wave = tid >> 5;
    const int wr = wave >> 2, wc = wave & 3;
    const int r0 = blockIdx.x * 64 + wr * 32;
    const int m0 = blockIdx.y * 128 + wc * 32;
    if (r0 >= N || m0 >= M) return;   // wave-uniform: EXEC stays all-ones

    v8f c00 = {0.f,0.f,0.f,0.f,0.f,0.f,0.f,0.f};
    v8f c01 = c00, c10 = c00, c11 = c00;

    const int arow0 = r0 + (lane & 15);
    const int brow0 = m0 + (lane & 15);

    for (int k0 = 0; k0 < K; k0 += 32) {
        v16h a0 = load_frag16(A,  arow0,      K, k0, lane);
        v16h a1 = load_frag16(A,  arow0 + 16, K, k0, lane);
        v16h b0 = load_frag16(Bt, brow0,      K, k0, lane);
        v16h b1 = load_frag16(Bt, brow0 + 16, K, k0, lane);
        c00 = __builtin_amdgcn_wmma_f32_16x16x32_f16(false, a0, false, b0, (short)0, c00, false, false);
        c01 = __builtin_amdgcn_wmma_f32_16x16x32_f16(false, a0, false, b1, (short)0, c01, false, false);
        c10 = __builtin_amdgcn_wmma_f32_16x16x32_f16(false, a1, false, b0, (short)0, c10, false, false);
        c11 = __builtin_amdgcn_wmma_f32_16x16x32_f16(false, a1, false, b1, (short)0, c11, false, false);
    }

    // D layout per 16x16 tile: lane<16 -> rows +0..7, col +lane; lane>=16 -> rows +8..15, col +lane-16
    const int colb  = (lane & 15);
    const int rsub  = ((lane >> 4) << 3);
#pragma unroll
    for (int ti = 0; ti < 2; ++ti) {
#pragma unroll
        for (int tj = 0; tj < 2; ++tj) {
            const v8f acc = (ti == 0) ? (tj == 0 ? c00 : c01) : (tj == 0 ? c10 : c11);
            const int col   = m0 + tj * 16 + colb;
            const int rbase = r0 + ti * 16 + rsub;
            const float bv  = bias ? bias[col] : 0.0f;
#pragma unroll
            for (int r = 0; r < 8; ++r) {
                const int row = rbase + r;
                float v = acc[r] + bv;
                if (GELU) v = 0.5f * v * (1.0f + erff(v * 0.70710678118654752f));
                const size_t idx = (size_t)row * M + col;
                if (OMODE == 0)      Cf32[idx] = v;
                else if (OMODE == 1) Cf32[idx] += v;
                else                 Cf16[idx] = (half_t)v;
            }
        }
    }
}

// ---------------------------------------------------------------------------
// LayerNorm over channel dim (C = CPL*32), one wave per pixel.
// ---------------------------------------------------------------------------
template <int CPL, bool OUT_F16>
__global__ __launch_bounds__(256) void layernorm_kernel(
    const float* __restrict__ x, const float* __restrict__ w,
    const float* __restrict__ b, half_t* __restrict__ outh,
    float* __restrict__ outf, int npix)
{
    constexpr int C = CPL * 32;
    const int lane = threadIdx.x & 31;
    const int wave = threadIdx.x >> 5;
    const int pix  = blockIdx.x * (blockDim.x >> 5) + wave;
    if (pix >= npix) return;

    const float* xp = x + (size_t)pix * C;
    float v[CPL];
    float s = 0.f;
#pragma unroll
    for (int i = 0; i < CPL; ++i) { v[i] = xp[lane + i * 32]; s += v[i]; }
    const float mu = wave_sum(s) * (1.0f / C);
    float var = 0.f;
#pragma unroll
    for (int i = 0; i < CPL; ++i) { float d = v[i] - mu; var += d * d; }
    var = wave_sum(var) * (1.0f / C);
    const float rstd = rsqrtf(var + 1e-5f);
#pragma unroll
    for (int i = 0; i < CPL; ++i) {
        const int c = lane + i * 32;
        const float o = (v[i] - mu) * rstd * w[c] + b[c];
        if (OUT_F16) outh[(size_t)pix * C + c] = (half_t)o;
        else         outf[(size_t)pix * C + c] = o;
    }
}

// ---------------------------------------------------------------------------
// Neighborhood attention: one wave per (pixel, head). head_dim == 32 == wave32.
// qkv layout per pixel: [q(128) | k(128) | v(128)].
// ---------------------------------------------------------------------------
__global__ __launch_bounds__(256) void nat_attention_kernel(
    const float* __restrict__ qkv, const float* __restrict__ rpb,
    half_t* __restrict__ outh)
{
    const int lane = threadIdx.x & 31;
    const int wave = threadIdx.x >> 5;
    const int unit = blockIdx.x * (blockDim.x >> 5) + wave;
    if (unit >= NPIX * NHEAD) return;

    const int pix  = unit >> 2;
    const int head = unit & 3;
    const int bimg = pix / (H_IMG * W_IMG);
    const int rem  = pix - bimg * H_IMG * W_IMG;
    const int i    = rem / W_IMG;
    const int j    = rem - i * W_IMG;

    int sr = i - KWIN / 2; sr = sr < 0 ? 0 : (sr > H_IMG - KWIN ? H_IMG - KWIN : sr);
    int sc = j - KWIN / 2; sc = sc < 0 ? 0 : (sc > W_IMG - KWIN ? W_IMG - KWIN : sc);

    const float scale = 0.17677669529663689f;  // 1/sqrt(32)
    const float q = qkv[(size_t)pix * 384 + head * HDIM + lane] * scale;
    const float* rpb_h = rpb + head * RPB_D * RPB_D;

    float sc0 = -1e30f, sc1 = -1e30f;
    for (int n = 0; n < NNB; ++n) {
        const int ki = n / KWIN, kj = n - ki * KWIN;
        const int ni = sr + ki, nj = sc + kj;
        const int npix = (bimg * H_IMG + ni) * W_IMG + nj;
        const float kv = qkv[(size_t)npix * 384 + C_DIM + head * HDIM + lane];
        float p = wave_sum(q * kv);
        p += rpb_h[(ni - i + KWIN - 1) * RPB_D + (nj - j + KWIN - 1)];
        if (n < 32) { if (lane == n)      sc0 = p; }
        else        { if (lane == n - 32) sc1 = p; }
    }

    const float m   = wave_max(fmaxf(sc0, sc1));
    const float e0  = __expf(sc0 - m);
    const float e1  = __expf(sc1 - m);     // -inf-ish -> 0 for invalid slots
    const float inv = 1.0f / wave_sum(e0 + e1);

    float acc = 0.f;
    for (int n = 0; n < NNB; ++n) {
        const int ki = n / KWIN, kj = n - ki * KWIN;
        const int ni = sr + ki, nj = sc + kj;
        const int npix = (bimg * H_IMG + ni) * W_IMG + nj;
        const float ev = (n < 32) ? __shfl(e0, n, 32) : __shfl(e1, n - 32, 32);
        const float vv = qkv[(size_t)npix * 384 + 2 * C_DIM + head * HDIM + lane];
        acc += ev * inv * vv;
    }
    outh[(size_t)pix * C_DIM + head * HDIM + lane] = (half_t)acc;
}

// ---------------------------------------------------------------------------
// im2col for 3x3 stride-2 pad-1 conv, f32 -> f16. One block per (out-pixel, tap).
// ---------------------------------------------------------------------------
__global__ __launch_bounds__(128) void im2col_kernel(
    const float* __restrict__ x, half_t* __restrict__ col)
{
    const int tap = blockIdx.x % 9;
    const int op  = blockIdx.x / 9;
    const int c   = threadIdx.x;
    const int bimg = op / (OH * OW);
    const int rem  = op - bimg * OH * OW;
    const int oi = rem / OW, oj = rem - oi * OW;
    const int di = tap / 3, dj = tap - di * 3;
    const int ii = oi * 2 - 1 + di;
    const int jj = oj * 2 - 1 + dj;
    float v = 0.f;
    if (ii >= 0 && ii < H_IMG && jj >= 0 && jj < W_IMG)
        v = x[(((size_t)(bimg * H_IMG + ii)) * W_IMG + jj) * C_DIM + c];
    col[(size_t)op * KCONV + tap * C_DIM + c] = (half_t)v;
}

// ---------------------------------------------------------------------------
// Weight convert + transpose: w[K x M] f32 -> wt[M x K] f16.
// ---------------------------------------------------------------------------
__global__ __launch_bounds__(256) void convert_transpose_kernel(
    const float* __restrict__ w, half_t* __restrict__ wt, int K, int M)
{
    const int idx = blockIdx.x * blockDim.x + threadIdx.x;
    if (idx >= K * M) return;
    const int m = idx / K;
    const int k = idx - m * K;
    wt[(size_t)m * K + k] = (half_t)w[(size_t)k * M + m];
}

// ---------------------------------------------------------------------------
// Host launcher
// ---------------------------------------------------------------------------
extern "C" void kernel_launch(void* const* d_in, const int* in_sizes, int n_in,
                              void* d_out, int out_size, void* d_ws, size_t ws_size,
                              hipStream_t stream)
{
    const float* x_in    = (const float*)d_in[0];
    const float* norm1_w = (const float*)d_in[1];
    const float* norm1_b = (const float*)d_in[2];
    const float* qkv_w   = (const float*)d_in[3];
    const float* qkv_b   = (const float*)d_in[4];
    const float* rpb     = (const float*)d_in[5];
    const float* proj_w  = (const float*)d_in[6];
    const float* proj_b  = (const float*)d_in[7];
    const float* norm2_w = (const float*)d_in[8];
    const float* norm2_b = (const float*)d_in[9];
    const float* fc1_w   = (const float*)d_in[10];
    const float* fc1_b   = (const float*)d_in[11];
    const float* fc2_w   = (const float*)d_in[12];
    const float* fc2_b   = (const float*)d_in[13];
    const float* conv_w  = (const float*)d_in[14];
    const float* dnorm_w = (const float*)d_in[15];
    const float* dnorm_b = (const float*)d_in[16];
    float* out = (float*)d_out;

    char* ws = (char*)d_ws;
    size_t off = 0;
    auto carve = [&](size_t bytes) -> void* {
        void* p = ws + off;
        off += (bytes + 255) & ~(size_t)255;
        return p;
    };

    float*  x_cur  = (float*)  carve((size_t)NPIX * C_DIM * 4);
    half_t* h16    = (half_t*) carve((size_t)NPIX * C_DIM * 2);
    float*  qkv32  = (float*)  carve((size_t)NPIX * 3 * C_DIM * 4);
    half_t* attn16 = (half_t*) carve((size_t)NPIX * C_DIM * 2);
    half_t* mlp16  = (half_t*) carve((size_t)NPIX * HID * 2);
    half_t* col16  = (half_t*) carve((size_t)NOUT * KCONV * 2);
    float*  conv32 = (float*)  carve((size_t)NOUT * MCONV * 4);
    half_t* qkvwt  = (half_t*) carve((size_t)2 * 3 * C_DIM * C_DIM * 2);
    half_t* projwt = (half_t*) carve((size_t)2 * C_DIM * C_DIM * 2);
    half_t* fc1wt  = (half_t*) carve((size_t)2 * HID * C_DIM * 2);
    half_t* fc2wt  = (half_t*) carve((size_t)2 * C_DIM * HID * 2);
    half_t* convwt = (half_t*) carve((size_t)MCONV * KCONV * 2);

    // x_cur = x (inputs stay immutable)
    hipMemcpyAsync(x_cur, x_in, (size_t)NPIX * C_DIM * 4,
                   hipMemcpyDeviceToDevice, stream);

    // Weight convert + transpose (f32 KxM -> f16 MxK)
    auto cvt = [&](const float* w, half_t* wt, int K, int M) {
        int n = K * M;
        convert_transpose_kernel<<<(n + 255) / 256, 256, 0, stream>>>(w, wt, K, M);
    };
    for (int l = 0; l < 2; ++l) {
        cvt(qkv_w  + (size_t)l * C_DIM * 3 * C_DIM, qkvwt  + (size_t)l * 3 * C_DIM * C_DIM, C_DIM, 3 * C_DIM);
        cvt(proj_w + (size_t)l * C_DIM * C_DIM,     projwt + (size_t)l * C_DIM * C_DIM,     C_DIM, C_DIM);
        cvt(fc1_w  + (size_t)l * C_DIM * HID,       fc1wt  + (size_t)l * HID * C_DIM,       C_DIM, HID);
        cvt(fc2_w  + (size_t)l * HID * C_DIM,       fc2wt  + (size_t)l * C_DIM * HID,       HID,   C_DIM);
    }
    cvt(conv_w, convwt, KCONV, MCONV);   // (3,3,128,256) flat == (1152, 256)

    const dim3 blk(256);
    auto gemm_grid = [](int N, int M) { return dim3((N + 63) / 64, M / 128); };

    for (int l = 0; l < 2; ++l) {
        // LN1 -> f16
        layernorm_kernel<4, true><<<NPIX / 8, blk, 0, stream>>>(
            x_cur, norm1_w + l * C_DIM, norm1_b + l * C_DIM, h16, nullptr, NPIX);
        // QKV GEMM: 6272x128 @ 128x384 (+bias) -> f32
        gemm_wmma_kernel<false, 0><<<gemm_grid(NPIX, 3 * C_DIM), blk, 0, stream>>>(
            h16, qkvwt + (size_t)l * 3 * C_DIM * C_DIM, qkv_b + l * 3 * C_DIM,
            qkv32, nullptr, NPIX, C_DIM, 3 * C_DIM);
        // Neighborhood attention -> f16
        nat_attention_kernel<<<(NPIX * NHEAD) / 8, blk, 0, stream>>>(
            qkv32, rpb + (size_t)l * NHEAD * RPB_D * RPB_D, attn16);
        // Proj GEMM, residual accumulate into x_cur
        gemm_wmma_kernel<false, 1><<<gemm_grid(NPIX, C_DIM), blk, 0, stream>>>(
            attn16, projwt + (size_t)l * C_DIM * C_DIM, proj_b + l * C_DIM,
            x_cur, nullptr, NPIX, C_DIM, C_DIM);
        // LN2 -> f16
        layernorm_kernel<4, true><<<NPIX / 8, blk, 0, stream>>>(
            x_cur, norm2_w + l * C_DIM, norm2_b + l * C_DIM, h16, nullptr, NPIX);
        // FC1 GEMM + GELU -> f16
        gemm_wmma_kernel<true, 2><<<gemm_grid(NPIX, HID), blk, 0, stream>>>(
            h16, fc1wt + (size_t)l * HID * C_DIM, fc1_b + l * HID,
            nullptr, mlp16, NPIX, C_DIM, HID);
        // FC2 GEMM, residual accumulate into x_cur
        gemm_wmma_kernel<false, 1><<<gemm_grid(NPIX, C_DIM), blk, 0, stream>>>(
            mlp16, fc2wt + (size_t)l * C_DIM * HID, fc2_b + l * C_DIM,
            x_cur, nullptr, NPIX, HID, C_DIM);
    }

    // Downsample conv as im2col + GEMM (no bias), then dnorm LN -> out[0 : 401408]
    im2col_kernel<<<NOUT * 9, 128, 0, stream>>>(x_cur, col16);
    gemm_wmma_kernel<false, 0><<<gemm_grid(NOUT, MCONV), blk, 0, stream>>>(
        col16, convwt, nullptr, conv32, nullptr, NOUT, KCONV, MCONV);
    layernorm_kernel<8, false><<<NOUT / 8, blk, 0, stream>>>(
        conv32, dnorm_w, dnorm_b, nullptr, out, NOUT);

    // Second tuple element: x after both layers
    hipMemcpyAsync(out + (size_t)NOUT * MCONV, x_cur, (size_t)NPIX * C_DIM * 4,
                   hipMemcpyDeviceToDevice, stream);
}
